// AttnBlockpp_14766097564201
// MI455X (gfx1250) — matrix operand order repaired and checked by reading, not compiled, not run on hardware
//
#include <hip/hip_runtime.h>
#include <hip/hip_bf16.h>

typedef __bf16 bf16;
typedef __attribute__((ext_vector_type(8)))  __bf16 v8bf;
typedef __attribute__((ext_vector_type(16))) __bf16 v16bf;
typedef __attribute__((ext_vector_type(8)))  float  v8f;
typedef __attribute__((ext_vector_type(4)))  float  v4f;

#define GAS __attribute__((address_space(1)))

#define BATCH 8
#define HW    4096          // 64*64
#define CH    256
#define NPIX  32768         // BATCH*HW
#define GRP   32
#define CPG   8             // CH/GRP

union V16U { v16bf v; v8bf h[2]; };

__device__ __forceinline__ unsigned short f2bf(float f) {
  unsigned u = __builtin_bit_cast(unsigned, f);
  unsigned r = (u + 0x7FFFu + ((u >> 16) & 1u)) >> 16;
  return (unsigned short)r;
}

__device__ __forceinline__ v8f vzero8() {
  v8f z;
#pragma unroll
  for (int i = 0; i < 8; ++i) z[i] = 0.0f;
  return z;
}

// ---- global-memory (addrspace 1) helpers: force global_load_*/global_store_* ----
__device__ __forceinline__ void gstore_u16(unsigned short* p, unsigned short v) {
  *(GAS unsigned short*)p = v;
}
__device__ __forceinline__ float gload_f32(const float* p) {
  return *(const GAS float*)p;
}
__device__ __forceinline__ void gstore_f32(float* p, float v) {
  *(GAS float*)p = v;
}
__device__ __forceinline__ v4f gload_v4f(const float* p) {
  return *(const GAS v4f*)p;
}

// A-operand tile (16x32 bf16, row-major src, row stride ld elements), global mem.
// Per ISA: lanes0-15 row=lane, K=0..7 & 16..23; lanes16-31 row=lane-16, K=8..15 & 24..31.
__device__ __forceinline__ v16bf load_a16_g(const bf16* base, int ld, int lane) {
  int row = lane & 15;
  int kb  = (lane & 16) ? 8 : 0;
  const bf16* p = base + (size_t)row * ld + kb;
  V16U u;
  u.h[0] = *(const GAS v8bf*)(p);
  u.h[1] = *(const GAS v8bf*)(p + 16);
  return u.v;
}

// Same A layout but from LDS (generic pointer -> ds_load).
__device__ __forceinline__ v16bf load_a16_s(const bf16* base, int ld, int lane) {
  int row = lane & 15;
  int kb  = (lane & 16) ? 8 : 0;
  const bf16* p = base + (size_t)row * ld + kb;
  V16U u;
  u.h[0] = *(const v8bf*)(p);
  u.h[1] = *(const v8bf*)(p + 16);
  return u.v;
}

// B-operand tile (32x16 bf16): element B[k][n] = base[n*ld + k], global mem.
// Per ISA: lanes0-15 (col n) K=0..7 & 16..23, lanes16-31 K=8..15 & 24..31.
__device__ __forceinline__ v16bf load_b16_g(const bf16* base, int ld, int lane) {
  int n  = lane & 15;
  int kb = (lane & 16) ? 8 : 0;
  const bf16* p = base + (size_t)n * ld + kb;
  V16U u;
  u.h[0] = *(const GAS v8bf*)(p);
  u.h[1] = *(const GAS v8bf*)(p + 16);
  return u.v;
}

__device__ __forceinline__ v8f wmma_bf16(v16bf a, v16bf b, v8f c) {
  return __builtin_amdgcn_wmma_f32_16x16x32_bf16(false, a, false, b,
                                                 (short)0, c, false, false);
}

// ---------------- 1. GroupNorm statistics ----------------
__global__ void gn_stats_kernel(const float* __restrict__ x,
                                float* __restrict__ mean,
                                float* __restrict__ rstd) {
  int bg = blockIdx.x;            // b*32+g
  int b  = bg >> 5;
  int g  = bg & 31;
  const float* xp = x + (size_t)b * HW * CH + g * CPG;
  float s = 0.f, ss = 0.f;
  for (int p = threadIdx.x; p < HW; p += 256) {
    const float* q = xp + (size_t)p * CH;
    v4f a = gload_v4f(q);
    v4f c = gload_v4f(q + 4);
#pragma unroll
    for (int i = 0; i < 4; ++i) {
      s += a[i] + c[i];
      ss += a[i] * a[i] + c[i] * c[i];
    }
  }
  __shared__ float sh1[256], sh2[256];
  sh1[threadIdx.x] = s; sh2[threadIdx.x] = ss;
  __syncthreads();
  for (int o = 128; o > 0; o >>= 1) {
    if (threadIdx.x < o) {
      sh1[threadIdx.x] += sh1[threadIdx.x + o];
      sh2[threadIdx.x] += sh2[threadIdx.x + o];
    }
    __syncthreads();
  }
  if (threadIdx.x == 0) {
    float m = sh1[0] * (1.0f / 32768.0f);
    float v = sh2[0] * (1.0f / 32768.0f) - m * m;
    gstore_f32(&mean[bg], m);
    gstore_f32(&rstd[bg], rsqrtf(v + 1e-6f));
  }
}

// ---------------- 2. GroupNorm apply -> h (bf16) ----------------
__global__ void gn_apply_kernel(const float* __restrict__ x,
                                const float* __restrict__ scale,
                                const float* __restrict__ bias,
                                const float* __restrict__ mean,
                                const float* __restrict__ rstd,
                                unsigned short* __restrict__ hbf) {
  int idx = blockIdx.x * 256 + threadIdx.x;   // 0 .. 1048575
  int pix = idx >> 5;
  int cg  = idx & 31;                          // group == channel-chunk
  int b   = pix >> 12;
  float m = gload_f32(&mean[b * 32 + cg]);
  float r = gload_f32(&rstd[b * 32 + cg]);
  const float* px = x + (size_t)pix * CH + cg * CPG;
  v4f a = gload_v4f(px);
  v4f c = gload_v4f(px + 4);
  unsigned short* ph = hbf + (size_t)pix * CH + cg * CPG;
  float v[8] = { a[0], a[1], a[2], a[3], c[0], c[1], c[2], c[3] };
#pragma unroll
  for (int i = 0; i < 8; ++i) {
    int ci = cg * CPG + i;
    float t = (v[i] - m) * r * gload_f32(&scale[ci]) + gload_f32(&bias[ci]);
    gstore_u16(&ph[i], f2bf(t));
  }
}

// ---------------- 3. Weight transpose fp32[c][d] -> bf16[d][c] ----------------
__global__ void wtrans_kernel(const float* __restrict__ w0, const float* __restrict__ w1,
                              const float* __restrict__ w2, const float* __restrict__ w3,
                              unsigned short* __restrict__ t0, unsigned short* __restrict__ t1,
                              unsigned short* __restrict__ t2, unsigned short* __restrict__ t3) {
  int widx = blockIdx.x >> 8;
  int d    = blockIdx.x & 255;
  int c    = threadIdx.x;
  const float* src = (widx == 0) ? w0 : (widx == 1) ? w1 : (widx == 2) ? w2 : w3;
  unsigned short* dst = (widx == 0) ? t0 : (widx == 1) ? t1 : (widx == 2) ? t2 : t3;
  gstore_u16(&dst[(size_t)d * CH + c], f2bf(gload_f32(&src[(size_t)c * CH + d])));
}

// ---------------- 4. QKV projection (WMMA bf16), target T compile-time ----------------
template <int T>
__device__ __forceinline__ void do_proj(const v16bf* A, const bf16* wt,
                                        const float* bias, unsigned short* dst,
                                        int row0, int lane) {
  int rofs = (lane & 16) ? 8 : 0;
  int colb = lane & 15;
#pragma unroll 1
  for (int nt = 0; nt < 16; ++nt) {
    v8f acc = vzero8();
#pragma unroll
    for (int kc = 0; kc < 8; ++kc)
      acc = wmma_bf16(A[kc], load_b16_g(wt + (size_t)(nt * 16) * CH + kc * 32, CH, lane), acc);
    int col = nt * 16 + colb;
    float bia = gload_f32(&bias[col]);
#pragma unroll
    for (int r = 0; r < 8; ++r) {
      int row = row0 + r + rofs;
      float val = acc[r] + bia;
      if (T == 0) {
        gstore_u16(&dst[(size_t)row * CH + col], f2bf(val * 0.0625f));  // fold C^-0.5
      } else if (T == 1) {
        gstore_u16(&dst[(size_t)row * CH + col], f2bf(val));
      } else {
        int b = row >> 12, pix = row & 4095;                            // vT: [b][c][pix]
        gstore_u16(&dst[((size_t)b * CH + col) * HW + pix], f2bf(val));
      }
    }
  }
}

__global__ void qkv_kernel(const unsigned short* __restrict__ hbf,
                           const unsigned short* __restrict__ wtq,
                           const unsigned short* __restrict__ wtk,
                           const unsigned short* __restrict__ wtv,
                           const float* __restrict__ bq,
                           const float* __restrict__ bk,
                           const float* __restrict__ bv,
                           unsigned short* __restrict__ qo,
                           unsigned short* __restrict__ ko,
                           unsigned short* __restrict__ vTo) {
  int lane = threadIdx.x & 31;
  int wave = threadIdx.x >> 5;
  int row0 = blockIdx.x * 128 + wave * 16;
  const bf16* h = (const bf16*)hbf;

  v16bf A[8];
#pragma unroll
  for (int kc = 0; kc < 8; ++kc)
    A[kc] = load_a16_g(h + (size_t)row0 * CH + kc * 32, CH, lane);

  do_proj<0>(A, (const bf16*)wtq, bq, qo, row0, lane);
  do_proj<1>(A, (const bf16*)wtk, bk, ko, row0, lane);
  do_proj<2>(A, (const bf16*)wtv, bv, vTo, row0, lane);
}

// ---------------- 5. Flash attention (WMMA + online softmax) ----------------
__global__ void attn_kernel(const unsigned short* __restrict__ qbf,
                            const unsigned short* __restrict__ kbf,
                            const unsigned short* __restrict__ vTbf,
                            unsigned short* __restrict__ attn) {
  __shared__ __align__(16) unsigned short pbuf[8][16 * 32];
  int lane = threadIdx.x & 31;
  int wave = threadIdx.x >> 5;
  int gt = blockIdx.x * 8 + wave;       // 0..2047
  int b  = gt >> 8;
  int qt = gt & 255;
  int row0 = qt * 16;

  const bf16* qb = (const bf16*)qbf + (size_t)b * HW * CH;
  const bf16* kb = (const bf16*)kbf + (size_t)b * HW * CH;
  const bf16* vb = (const bf16*)vTbf + (size_t)b * CH * HW;

  v16bf Aq[8];
#pragma unroll
  for (int kc = 0; kc < 8; ++kc)
    Aq[kc] = load_a16_g(qb + (size_t)row0 * CH + kc * 32, CH, lane);

  v8f acc[16];
#pragma unroll
  for (int nt = 0; nt < 16; ++nt) acc[nt] = vzero8();
  float m[8], l[8];
#pragma unroll
  for (int r = 0; r < 8; ++r) { m[r] = -1e30f; l[r] = 0.0f; }

  int rofs = (lane & 16) ? 8 : 0;
  int colb = lane & 15;
  unsigned short* pw = pbuf[wave];

  for (int j = 0; j < HW; j += 32) {
    if (j + 32 < HW) {
      __builtin_prefetch((const GAS bf16*)(kb + (size_t)(j + 32) * CH), 0, 3);
      __builtin_prefetch((const GAS bf16*)(vb + (size_t)colb * HW + j + 32), 0, 3);
    }
    v8f s0 = vzero8(), s1 = vzero8();
#pragma unroll
    for (int kc = 0; kc < 8; ++kc)
      s0 = wmma_bf16(Aq[kc], load_b16_g(kb + (size_t)j * CH + kc * 32, CH, lane), s0);
#pragma unroll
    for (int kc = 0; kc < 8; ++kc)
      s1 = wmma_bf16(Aq[kc], load_b16_g(kb + (size_t)(j + 16) * CH + kc * 32, CH, lane), s1);

    float alpha[8];
#pragma unroll
    for (int r = 0; r < 8; ++r) {
      float mx = fmaxf(s0[r], s1[r]);
#pragma unroll
      for (int o = 1; o < 16; o <<= 1) mx = fmaxf(mx, __shfl_xor(mx, o, 32));
      float mn = fmaxf(m[r], mx);
      alpha[r] = __expf(m[r] - mn);
      float p0 = __expf(s0[r] - mn);
      float p1 = __expf(s1[r] - mn);
      s0[r] = p0; s1[r] = p1;
      float rs = p0 + p1;
#pragma unroll
      for (int o = 1; o < 16; o <<= 1) rs += __shfl_xor(rs, o, 32);
      l[r] = l[r] * alpha[r] + rs;
      m[r] = mn;
    }
#pragma unroll
    for (int nt = 0; nt < 16; ++nt)
#pragma unroll
      for (int r = 0; r < 8; ++r) acc[nt][r] *= alpha[r];

    // C-layout -> row-major P tile in LDS (16x32 bf16)
#pragma unroll
    for (int r = 0; r < 8; ++r) {
      int row = r + rofs;
      pw[row * 32 + colb]      = f2bf(s0[r]);
      pw[row * 32 + 16 + colb] = f2bf(s1[r]);
    }
    asm volatile("s_wait_dscnt 0" ::: "memory");
    v16bf Ap = load_a16_s((const bf16*)pw, 32, lane);
#pragma unroll
    for (int nt = 0; nt < 16; ++nt)
      acc[nt] = wmma_bf16(Ap, load_b16_g(vb + (size_t)(nt * 16) * HW + j, HW, lane), acc[nt]);
  }

  float inv[8];
#pragma unroll
  for (int r = 0; r < 8; ++r) inv[r] = 1.0f / l[r];
#pragma unroll
  for (int nt = 0; nt < 16; ++nt) {
    int col = nt * 16 + colb;
#pragma unroll
    for (int r = 0; r < 8; ++r) {
      int row = row0 + r + rofs;
      gstore_u16(&attn[((size_t)b * HW + row) * CH + col], f2bf(acc[nt][r] * inv[r]));
    }
  }
}

// ---------------- 6. Output projection + residual ----------------
__global__ void outproj_kernel(const float* __restrict__ x,
                               const unsigned short* __restrict__ attn,
                               const unsigned short* __restrict__ wto,
                               const float* __restrict__ bo,
                               float* __restrict__ out) {
  int lane = threadIdx.x & 31;
  int wave = threadIdx.x >> 5;
  int row0 = blockIdx.x * 128 + wave * 16;
  const bf16* a = (const bf16*)attn;
  const bf16* wt = (const bf16*)wto;

  v16bf A[8];
#pragma unroll
  for (int kc = 0; kc < 8; ++kc)
    A[kc] = load_a16_g(a + (size_t)row0 * CH + kc * 32, CH, lane);

  int rofs = (lane & 16) ? 8 : 0;
  int colb = lane & 15;
#pragma unroll 1
  for (int nt = 0; nt < 16; ++nt) {
    v8f acc = vzero8();
#pragma unroll
    for (int kc = 0; kc < 8; ++kc)
      acc = wmma_bf16(A[kc], load_b16_g(wt + (size_t)(nt * 16) * CH + kc * 32, CH, lane), acc);
    int col = nt * 16 + colb;
    float bia = gload_f32(&bo[col]);
#pragma unroll
    for (int r = 0; r < 8; ++r) {
      int row = row0 + r + rofs;
      size_t off = (size_t)row * CH + col;
      gstore_f32(&out[off], gload_f32(&x[off]) + acc[r] + bia);
    }
  }
}

extern "C" void kernel_launch(void* const* d_in, const int* in_sizes, int n_in,
                              void* d_out, int out_size, void* d_ws, size_t ws_size,
                              hipStream_t stream) {
  const float* x   = (const float*)d_in[0];
  const float* gsc = (const float*)d_in[1];
  const float* gbi = (const float*)d_in[2];
  const float* Wq  = (const float*)d_in[3];
  const float* bq  = (const float*)d_in[4];
  const float* Wk  = (const float*)d_in[5];
  const float* bk  = (const float*)d_in[6];
  const float* Wv  = (const float*)d_in[7];
  const float* bv  = (const float*)d_in[8];
  const float* Wo  = (const float*)d_in[9];
  const float* bo  = (const float*)d_in[10];
  float* out = (float*)d_out;

  char* w = (char*)d_ws;
  float* mean = (float*)w;                       // 256 f
  float* rstd = (float*)(w + 1024);              // 256 f
  const size_t SZ = (size_t)NPIX * CH * sizeof(unsigned short);  // 16.78 MB
  unsigned short* hbf  = (unsigned short*)(w + 4096);
  unsigned short* qbf  = (unsigned short*)(w + 4096 + 1 * SZ);
  unsigned short* kbf  = (unsigned short*)(w + 4096 + 2 * SZ);
  unsigned short* vTbf = (unsigned short*)(w + 4096 + 3 * SZ);
  unsigned short* attn = (unsigned short*)(w + 4096 + 4 * SZ);
  unsigned short* wtq  = (unsigned short*)(w + 4096 + 5 * SZ);
  unsigned short* wtk  = wtq + (size_t)CH * CH;
  unsigned short* wtv  = wtk + (size_t)CH * CH;
  unsigned short* wto  = wtv + (size_t)CH * CH;

  gn_stats_kernel<<<BATCH * GRP, 256, 0, stream>>>(x, mean, rstd);
  gn_apply_kernel<<<4096, 256, 0, stream>>>(x, gsc, gbi, mean, rstd, hbf);
  wtrans_kernel<<<1024, 256, 0, stream>>>(Wq, Wk, Wv, Wo, wtq, wtk, wtv, wto);
  qkv_kernel<<<NPIX / 128, 256, 0, stream>>>(hbf, wtq, wtk, wtv, bq, bk, bv,
                                             qbf, kbf, vTbf);
  attn_kernel<<<(BATCH * HW / 16) / 8, 256, 0, stream>>>(qbf, kbf, vTbf, attn);
  outproj_kernel<<<NPIX / 128, 256, 0, stream>>>(x, attn, wto, bo, out);
}
